// TransformerCN_16312285790928
// MI455X (gfx1250) — matrix operand order repaired and checked by reading, not compiled, not run on hardware
//
#include <hip/hip_runtime.h>
#include <math.h>

// ---------------------------------------------------------------------------
// Types for WMMA operands (gfx1250, wave32)
// ---------------------------------------------------------------------------
typedef __attribute__((ext_vector_type(16))) __bf16          v16bf;
typedef __attribute__((ext_vector_type(16))) unsigned short  v16u;
typedef __attribute__((ext_vector_type(8)))  float           v8f;

// ---------------------------------------------------------------------------
// Helpers
// ---------------------------------------------------------------------------
__device__ __forceinline__ unsigned short f2bf_rne(float f) {
    unsigned int u = __float_as_uint(f);
    unsigned int r = (u + 0x7FFFu + ((u >> 16) & 1u)) >> 16; // round-nearest-even
    return (unsigned short)r;
}

// order-preserving float atomic max (init buffer to -inf)
__device__ __forceinline__ void atomicMaxF(float* addr, float val) {
    int iv = __float_as_int(val);
    if (iv >= 0) atomicMax((int*)addr, iv);
    else         atomicMin((unsigned int*)addr, (unsigned int)iv);
}

// ---------------------------------------------------------------------------
// Elementwise utility kernels
// ---------------------------------------------------------------------------
__global__ void fill_f32(float* __restrict__ p, float v, int n) {
    int i = blockIdx.x * blockDim.x + threadIdx.x;
    if (i < n) p[i] = v;
}

__global__ void relu_f32(float* __restrict__ p, int n) {
    int i = blockIdx.x * blockDim.x + threadIdx.x;
    if (i < n) { float v = p[i]; p[i] = v > 0.0f ? v : 0.0f; }
}

// ---------------------------------------------------------------------------
// Pack fp32 weight [fin x 128] into WMMA B-operand order (bf16):
//   Wp[(((kc*2 + khi)*128 + col) << 4) + i] = bf16(W[(kc*32 + khi*16 + i)*128 + col])
// so each lane reads 16 contiguous bf16 (32B) per k-step.
// ---------------------------------------------------------------------------
__global__ void pack_w_bf16(const float* __restrict__ W,
                            unsigned short* __restrict__ Wp, int fin) {
    int tid = blockIdx.x * blockDim.x + threadIdx.x;
    int total = fin * 128;
    if (tid >= total) return;
    int krow = tid >> 7;       // 0..fin-1
    int col  = tid & 127;
    int kc   = krow >> 5;
    int rem  = krow & 31;
    int khi  = rem >> 4;
    int i    = rem & 15;
    Wp[(((((kc << 1) + khi) * 128) + col) << 4) + i] = f2bf_rne(W[krow * 128 + col]);
}

// ---------------------------------------------------------------------------
// Fused WMMA GEMM for the four projections sharing the same A operand:
//   Q = X@Wq+bq, K = X@Wk+bk, V = X@Wv+bv, S = X@Ws+bs     (all [nrows x 128])
// One block = 16 rows x 128 cols; 8 waves, wave w owns column tile w for all
// four outputs. The A fragment is loaded once per k-step; the four B operands
// are loaded into DISTINCT registers so the loads clause together and the
// WMMAs can retire against partial load-counter waits.
// ---------------------------------------------------------------------------
__global__ __launch_bounds__(256) void gemm_qkvs_wmma(
    const float* __restrict__ X,
    const unsigned short* __restrict__ WpQ, const unsigned short* __restrict__ WpK,
    const unsigned short* __restrict__ WpV, const unsigned short* __restrict__ WpS,
    const float* __restrict__ bq, const float* __restrict__ bk,
    const float* __restrict__ bv, const float* __restrict__ bs,
    float* __restrict__ Qo, float* __restrict__ Ko,
    float* __restrict__ Vo, float* __restrict__ So,
    int nrows, int fin) {
    __shared__ unsigned short lx[16][136];     // bf16 X tile, padded rows

    const int row0 = blockIdx.x << 4;
    for (int i = threadIdx.x; i < (fin << 4); i += blockDim.x) {
        int r = i / fin, k = i - r * fin;
        int gr = row0 + r;
        float v = (gr < nrows) ? X[gr * fin + k] : 0.0f;
        lx[r][k] = f2bf_rne(v);
    }
    __syncthreads();

    const int lane = threadIdx.x & 31;
    const int wv   = threadIdx.x >> 5;    // 0..7 -> column tile
    const int mn   = lane & 15;           // A: M index / B,C,D: N index
    const int khi  = lane >> 4;           // K-half selector
    const int col  = (wv << 4) + mn;

    v8f aq, ak, av, as;
    {
        float b0 = bq[col], b1 = bk[col], b2 = bv[col], b3 = bs[col];
#pragma unroll
        for (int r = 0; r < 8; ++r) { aq[r] = b0; ak[r] = b1; av[r] = b2; as[r] = b3; }
    }

    for (int k0 = 0; k0 < fin; k0 += 32) {
        // A operand: lane holds K = {16*khi..16*khi+15}
        v16u au;
        const unsigned short* ap0 = &lx[mn][k0 + (khi << 4)];
#pragma unroll
        for (int i = 0; i < 16; ++i) { au[i] = ap0[i]; }
        const v16bf a = __builtin_bit_cast(v16bf, au);

        // B operands: 16 contiguous bf16 (32B) per matrix; distinct registers
        // so all 8 b128 loads can issue as one clause.
        const int boff = ((((k0 >> 5) << 1) + khi) * 128 + col) << 4;
        const v16u bu0 = *reinterpret_cast<const v16u*>(WpQ + boff);
        const v16u bu1 = *reinterpret_cast<const v16u*>(WpK + boff);
        const v16u bu2 = *reinterpret_cast<const v16u*>(WpV + boff);
        const v16u bu3 = *reinterpret_cast<const v16u*>(WpS + boff);

        aq = __builtin_amdgcn_wmma_f32_16x16x32_bf16(
            false, a, false, __builtin_bit_cast(v16bf, bu0), (short)0, aq, false, false);
        ak = __builtin_amdgcn_wmma_f32_16x16x32_bf16(
            false, a, false, __builtin_bit_cast(v16bf, bu1), (short)0, ak, false, false);
        av = __builtin_amdgcn_wmma_f32_16x16x32_bf16(
            false, a, false, __builtin_bit_cast(v16bf, bu2), (short)0, av, false, false);
        as = __builtin_amdgcn_wmma_f32_16x16x32_bf16(
            false, a, false, __builtin_bit_cast(v16bf, bu3), (short)0, as, false, false);
    }

    // C/D layout: element r -> M = r + 8*khi, N = col
#pragma unroll
    for (int r = 0; r < 8; ++r) {
        int orow = row0 + (khi << 3) + r;
        if (orow < nrows) {
            Qo[orow * 128 + col] = aq[r];
            Ko[orow * 128 + col] = ak[r];
            Vo[orow * 128 + col] = av[r];
            So[orow * 128 + col] = as[r];
        }
    }
}

// ---------------------------------------------------------------------------
// Edge pass A: raw attention logits + per-(dst,head) running max.
// One wave per edge; lane = channel within head (C = 32).
// ---------------------------------------------------------------------------
__global__ __launch_bounds__(256) void edge_alpha(
    const float* __restrict__ Q, const float* __restrict__ Kf,
    const float* __restrict__ ea, const float* __restrict__ we,
    const int* __restrict__ src, const int* __restrict__ dst,
    float* __restrict__ alpha, float* __restrict__ mbuf, int nE) {
    __shared__ float lwe[512];
    for (int i = threadIdx.x; i < 512; i += blockDim.x) lwe[i] = we[i];
    __syncthreads();

    const int lane = threadIdx.x & 31;
    const int e = blockIdx.x * (blockDim.x >> 5) + (threadIdx.x >> 5);
    if (e >= nE) return;
    const int s = src[e], d = dst[e];
    const float ea0 = ea[e * 4 + 0], ea1 = ea[e * 4 + 1];
    const float ea2 = ea[e * 4 + 2], ea3 = ea[e * 4 + 3];

#pragma unroll
    for (int h = 0; h < 4; ++h) {
        int c = (h << 5) + lane;
        float ep = ea0 * lwe[c] + ea1 * lwe[128 + c] + ea2 * lwe[256 + c] + ea3 * lwe[384 + c];
        float p = Q[d * 128 + c] * (Kf[s * 128 + c] + ep);
#pragma unroll
        for (int off = 16; off; off >>= 1) p += __shfl_xor(p, off, 32);
        if (lane == 0) {
            float a = p * 0.17677669529663687f;   // 1/sqrt(32)
            alpha[e * 4 + h] = a;
            atomicMaxF(&mbuf[d * 4 + h], a);
        }
    }
}

// ---------------------------------------------------------------------------
// Edge pass B: exp(alpha - max) and per-(dst,head) sum. Thread per (edge,head).
// ---------------------------------------------------------------------------
__global__ void edge_exp(float* __restrict__ alpha, const float* __restrict__ mbuf,
                         float* __restrict__ sbuf, const int* __restrict__ dst, int nE) {
    int tid = blockIdx.x * blockDim.x + threadIdx.x;
    if (tid >= nE * 4) return;
    int e = tid >> 2, h = tid & 3;
    int d = dst[e];
    float a = __expf(alpha[tid] - mbuf[d * 4 + h]);
    alpha[tid] = a;
    atomicAdd(&sbuf[d * 4 + h], a);
}

// ---------------------------------------------------------------------------
// Edge pass C: scatter messages (v[src]+e)*softmax into OUT (pre-loaded with skip).
// ---------------------------------------------------------------------------
__global__ __launch_bounds__(256) void edge_msg(
    const float* __restrict__ Vf, const float* __restrict__ ea,
    const float* __restrict__ we, const int* __restrict__ src,
    const int* __restrict__ dst, const float* __restrict__ alpha,
    const float* __restrict__ sbuf, float* __restrict__ OUT, int nE) {
    __shared__ float lwe[512];
    for (int i = threadIdx.x; i < 512; i += blockDim.x) lwe[i] = we[i];
    __syncthreads();

    const int lane = threadIdx.x & 31;
    const int e = blockIdx.x * (blockDim.x >> 5) + (threadIdx.x >> 5);
    if (e >= nE) return;
    const int s = src[e], d = dst[e];
    const float ea0 = ea[e * 4 + 0], ea1 = ea[e * 4 + 1];
    const float ea2 = ea[e * 4 + 2], ea3 = ea[e * 4 + 3];

#pragma unroll
    for (int h = 0; h < 4; ++h) {
        int c = (h << 5) + lane;
        float coef = alpha[e * 4 + h] / (sbuf[d * 4 + h] + 1e-16f);
        float ep = ea0 * lwe[c] + ea1 * lwe[128 + c] + ea2 * lwe[256 + c] + ea3 * lwe[384 + c];
        atomicAdd(&OUT[d * 128 + c], (Vf[s * 128 + c] + ep) * coef);
    }
}

// ---------------------------------------------------------------------------
// Global mean pool accumulation + final linear 128->1
// ---------------------------------------------------------------------------
__global__ void pool_accum(const float* __restrict__ H, const int* __restrict__ batch,
                           float* __restrict__ pooled, float* __restrict__ cnt, int n) {
    int tid = blockIdx.x * blockDim.x + threadIdx.x;
    if (tid >= n * 128) return;
    int node = tid >> 7, c = tid & 127;
    int b = batch[node];
    atomicAdd(&pooled[b * 128 + c], H[tid]);
    if (c == 0) atomicAdd(&cnt[b], 1.0f);
}

__global__ void final_linear(const float* __restrict__ pooled, const float* __restrict__ cnt,
                             const float* __restrict__ wlin, const float* __restrict__ blin,
                             float* __restrict__ out, int gcount) {
    int lane = threadIdx.x & 31;
    int g = blockIdx.x * (blockDim.x >> 5) + (threadIdx.x >> 5);
    if (g >= gcount) return;
    float p = 0.0f;
#pragma unroll
    for (int t = 0; t < 4; ++t) {
        int c = (t << 5) + lane;
        p += pooled[g * 128 + c] * wlin[c];
    }
#pragma unroll
    for (int off = 16; off; off >>= 1) p += __shfl_xor(p, off, 32);
    if (lane == 0) {
        float cn = cnt[g]; cn = cn > 1.0f ? cn : 1.0f;
        out[g] = p / cn + blin[0];
    }
}

// ---------------------------------------------------------------------------
// Host launcher
// ---------------------------------------------------------------------------
extern "C" void kernel_launch(void* const* d_in, const int* in_sizes, int n_in,
                              void* d_out, int out_size, void* d_ws, size_t ws_size,
                              hipStream_t stream) {
    (void)n_in; (void)ws_size;
    const int N = in_sizes[3];          // nodes (batch vector length)
    const int E = in_sizes[1] / 2;      // edges
    const int G = out_size;             // graphs

    const float* x     = (const float*)d_in[0];
    const int*   ei    = (const int*)d_in[1];
    const int*   src   = ei;
    const int*   dst   = ei + E;
    const float* ea    = (const float*)d_in[2];
    const int*   batch = (const int*)d_in[3];
    const float* wlin  = (const float*)d_in[31];
    const float* blin  = (const float*)d_in[32];

    // workspace carve-out (256B aligned chunks)
    char* ws = (char*)d_ws;
    size_t off = 0;
    auto carve = [&](size_t bytes) -> char* {
        char* p = ws + off;
        off += (bytes + 255) & ~(size_t)255;
        return p;
    };
    float* Qb    = (float*)carve((size_t)N * 128 * 4);
    float* Kb    = (float*)carve((size_t)N * 128 * 4);
    float* Vb    = (float*)carve((size_t)N * 128 * 4);
    float* OUTA  = (float*)carve((size_t)N * 128 * 4);
    float* OUTB  = (float*)carve((size_t)N * 128 * 4);
    float* ALPHA = (float*)carve((size_t)E * 4 * 4);
    float* Mb    = (float*)carve((size_t)N * 4 * 4);
    float* Sb    = (float*)carve((size_t)N * 4 * 4);
    float* POOL  = (float*)carve((size_t)G * 128 * 4);
    float* CNT   = (float*)carve((size_t)G * 4);
    unsigned short* WpQ = (unsigned short*)carve(128 * 128 * 2);
    unsigned short* WpK = (unsigned short*)carve(128 * 128 * 2);
    unsigned short* WpV = (unsigned short*)carve(128 * 128 * 2);
    unsigned short* WpS = (unsigned short*)carve(128 * 128 * 2);

    const float* Xcur = x;
    int fin = in_sizes[0] / N;                      // 64 for layer 1
    const int gemm_grid  = (N + 15) / 16;
    const int ewave_grid = (E + 7) / 8;             // 8 waves per 256-thread block

    for (int l = 0; l < 3; ++l) {
        const int base = 4 + 9 * l;
        const float* wq  = (const float*)d_in[base + 0];
        const float* bq  = (const float*)d_in[base + 1];
        const float* wk  = (const float*)d_in[base + 2];
        const float* bk  = (const float*)d_in[base + 3];
        const float* wv  = (const float*)d_in[base + 4];
        const float* bv  = (const float*)d_in[base + 5];
        const float* wel = (const float*)d_in[base + 6];
        const float* wsk = (const float*)d_in[base + 7];
        const float* bs  = (const float*)d_in[base + 8];

        const int pk_grid = (fin * 128 + 255) / 256;
        pack_w_bf16<<<pk_grid, 256, 0, stream>>>(wq,  WpQ, fin);
        pack_w_bf16<<<pk_grid, 256, 0, stream>>>(wk,  WpK, fin);
        pack_w_bf16<<<pk_grid, 256, 0, stream>>>(wv,  WpV, fin);
        pack_w_bf16<<<pk_grid, 256, 0, stream>>>(wsk, WpS, fin);

        float* OUT = (l & 1) ? OUTB : OUTA;
        gemm_qkvs_wmma<<<gemm_grid, 256, 0, stream>>>(
            Xcur, WpQ, WpK, WpV, WpS, bq, bk, bv, bs,
            Qb, Kb, Vb, OUT, N, fin);

        fill_f32<<<(N * 4 + 255) / 256, 256, 0, stream>>>(Mb, -INFINITY, N * 4);
        fill_f32<<<(N * 4 + 255) / 256, 256, 0, stream>>>(Sb, 0.0f,      N * 4);

        edge_alpha<<<ewave_grid, 256, 0, stream>>>(Qb, Kb, ea, wel, src, dst, ALPHA, Mb, E);
        edge_exp<<<(E * 4 + 255) / 256, 256, 0, stream>>>(ALPHA, Mb, Sb, dst, E);
        edge_msg<<<ewave_grid, 256, 0, stream>>>(Vb, ea, wel, src, dst, ALPHA, Sb, OUT, E);

        if (l < 2) relu_f32<<<(N * 128 + 255) / 256, 256, 0, stream>>>(OUT, N * 128);
        Xcur = OUT;
        fin = 128;
    }

    fill_f32<<<(G * 128 + 255) / 256, 256, 0, stream>>>(POOL, 0.0f, G * 128);
    fill_f32<<<(G + 255) / 256, 256, 0, stream>>>(CNT, 0.0f, G);
    pool_accum<<<(N * 128 + 255) / 256, 256, 0, stream>>>(Xcur, batch, POOL, CNT, N);
    final_linear<<<(G + 7) / 8, 256, 0, stream>>>(POOL, CNT, wlin, blin, (float*)d_out, G);
}
